// HypergraphComputation_16080357556288
// MI455X (gfx1250) — matrix operand order repaired and checked by reading, not compiled, not run on hardware
//
#include <hip/hip_runtime.h>
#include <hip/hip_bf16.h>

// ---------------- CDNA5 WMMA types ----------------
typedef __bf16 v16bf __attribute__((ext_vector_type(16)));
typedef float  v8f   __attribute__((ext_vector_type(8)));
typedef int    v4i   __attribute__((ext_vector_type(4)));

#define BNUM 4
#define CDIM 256
#define NPIX 1024
#define NCTX 2048
#define VBLK 3072   /* N + Nc per batch block */
#define NTGT 4096   /* B*N  (edge count E, also target rows) */
#define NCTXALL 8192
#define NALL 12288  /* B*(N+Nc) = V */

// ---- gfx1250 async global->LDS path (guarded; fallback = load + ds_store) ----
#if defined(__has_builtin)
#if __has_builtin(__builtin_amdgcn_global_load_async_to_lds_b128)
#define HAVE_ASYNC_LDS 1
#endif
#endif

typedef __attribute__((address_space(1))) v4i* as1_v4i;
typedef __attribute__((address_space(3))) v4i* as3_v4i;

__device__ __forceinline__ void copy16_g2l(const __bf16* g, __bf16* l) {
#ifdef HAVE_ASYNC_LDS
  __builtin_amdgcn_global_load_async_to_lds_b128((as1_v4i)(g), (as3_v4i)(l), 0, 0);
#else
  *(v4i*)l = *(const v4i*)g;
#endif
}

__device__ __forceinline__ void stage_fence() {
#ifdef HAVE_ASYNC_LDS
#if __has_builtin(__builtin_amdgcn_s_wait_asynccnt)
  __builtin_amdgcn_s_wait_asynccnt(0);
#else
  asm volatile("s_wait_asynccnt 0" ::: "memory");
#endif
#endif
  __syncthreads();
}

// Stage a 64-row x 32-half B panel (rows of T starting at n0, halfs kabs..kabs+31)
// into LDS [64][32]. 128 threads: 2 threads/row, 32B each (2 x b128).
__device__ __forceinline__ void stage_b_panel(const __bf16* __restrict__ T, size_t ld,
                                              int n0, size_t kabs, __bf16* lds) {
  const int tid = threadIdx.x;
  const int row = tid >> 1;
  const int half = tid & 1;
  const __bf16* g = T + (size_t)(n0 + row) * ld + kabs + (half << 4);
  __bf16* l = lds + (row << 5) + (half << 4);
  copy16_g2l(g, l);
  copy16_g2l(g + 8, l + 8);
}

__device__ __forceinline__ v8f wmma_bf16(v16bf a, v16bf b, v8f c) {
  // (neg_a, A, neg_b, B, c_mod, C, reuse_a, reuse_b)
  return __builtin_amdgcn_wmma_f32_16x16x32_bf16(false, a, false, b, (short)0, c,
                                                 false, false);
}

// A fragment (16x32 bf16) from row-major A, tile origin (m0,k0).
// lane<16 -> M=lane, K {0..15}; lane>=16 -> M=lane-16, K {16..31} (contiguous, matches B frags)
__device__ __forceinline__ v16bf frag_a(const __bf16* A, int lda, int m0, int k0) {
  const int lane = threadIdx.x & 31;
  const __bf16* p = A + (size_t)(m0 + (lane & 15)) * lda + (k0 + ((lane >> 4) << 4));
  v16bf f;
#pragma unroll
  for (int i = 0; i < 8; ++i) f[i] = p[i];
#pragma unroll
  for (int i = 0; i < 8; ++i) f[8 + i] = p[8 + i];
  return f;
}

// B fragment (32x16 bf16) where B[k][n] = T[n][kabs+k], T row-major with leading dim ld.
__device__ __forceinline__ v16bf frag_bT(const __bf16* T, size_t ld, int n0, size_t kabs) {
  const int lane = threadIdx.x & 31;
  const __bf16* p = T + (size_t)(n0 + (lane & 15)) * ld + kabs + ((lane >> 4) << 4);
  v16bf f;
#pragma unroll
  for (int i = 0; i < 16; ++i) f[i] = p[i];
  return f;
}

// B fragment from LDS panel [64][32]: column n_local..n_local+15 of this K-step.
__device__ __forceinline__ v16bf frag_b_lds(const __bf16* P, int n_local) {
  const int lane = threadIdx.x & 31;
  const __bf16* p = P + ((n_local + (lane & 15)) << 5) + ((lane >> 4) << 4);
  v16bf f;
#pragma unroll
  for (int i = 0; i < 16; ++i) f[i] = p[i];
  return f;
}

// ---------------- Phase 0: NCHW -> row features + normalized bf16 rows ----------------
__global__ void row_prep(const float* __restrict__ Xt, const float* __restrict__ Xc1,
                         const float* __restrict__ Xc2, __bf16* __restrict__ XAbf,
                         __bf16* __restrict__ NRMbf) {
  const int row = blockIdx.x;     // X_all ordering: targets then contexts
  const int c = threadIdx.x;      // channel
  const float* src;
  int b, n;
  if (row < NTGT) {
    b = row >> 10; n = row & 1023; src = Xt;
  } else {
    const int cr = row - NTGT;
    b = cr >> 11;
    const int l = cr & 2047;
    n = l & 1023;
    src = (l >> 10) ? Xc2 : Xc1;
  }
  const float x = src[((size_t)(b * CDIM + c) << 10) + n];
  XAbf[(size_t)row * CDIM + c] = (__bf16)x;

  __shared__ float part[8];
  __shared__ float invn;
  float v = x * x;
#pragma unroll
  for (int off = 16; off; off >>= 1) v += __shfl_down(v, off, 32);
  if ((threadIdx.x & 31) == 0) part[threadIdx.x >> 5] = v;
  __syncthreads();
  if (threadIdx.x == 0) {
    float t = 0.f;
#pragma unroll
    for (int i = 0; i < 8; ++i) t += part[i];
    invn = 1.0f / fmaxf(sqrtf(t), 1e-8f);
  }
  __syncthreads();
  NRMbf[(size_t)row * CDIM + c] = (__bf16)(x * invn);
}

__global__ void cvt_w(const float* __restrict__ W, __bf16* __restrict__ Wb, int n) {
  const int i = blockIdx.x * blockDim.x + threadIdx.x;
  if (i < n) Wb[i] = (__bf16)W[i];
}

// ---------------- Phase 1: cosine-sim GEMM + threshold -> mask / maskT ----------------
// Block = 4 waves stacked in M (64 rows) x 64-col strip; shared B panel in LDS,
// double-buffered via async global->LDS. blocks/batch = 16 * 32 = 512.
__global__ void sim_mask(const __bf16* __restrict__ NRM, __bf16* __restrict__ mask,
                         __bf16* __restrict__ maskT) {
  __shared__ __bf16 bpan[2][64 * 32];
  const int blk = blockIdx.x;
  const int b = blk >> 9;            // 512 blocks per batch
  const int t = blk & 511;
  const int m0 = ((t >> 5) << 6) + ((threadIdx.x >> 5) << 4);  // 16 M-blocks * 64 + wave*16
  const int n0 = (t & 31) << 6;                                // 32 strips of 64 cols
  const __bf16* Abase = NRM + (size_t)(b * NPIX) * CDIM;           // tn rows
  const __bf16* Sbase = NRM + (size_t)(NTGT + b * NCTX) * CDIM;    // cn rows
  v8f acc[4] = {{}, {}, {}, {}};
  stage_b_panel(Sbase, CDIM, n0, 0, bpan[0]);
#pragma unroll
  for (int i = 0; i < 8; ++i) {          // K = 256 in steps of 32
    const int k0 = i << 5;
    stage_fence();                        // bpan[i&1] ready on all waves
    if (i + 1 < 8) stage_b_panel(Sbase, CDIM, n0, (size_t)(k0 + 32), bpan[(i + 1) & 1]);
    const v16bf a = frag_a(Abase, CDIM, m0, k0);
    const __bf16* P = bpan[i & 1];
#pragma unroll
    for (int tt = 0; tt < 4; ++tt) {
      const v16bf bb = frag_b_lds(P, tt << 4);
      acc[tt] = wmma_bf16(a, bb, acc[tt]);
    }
  }
  const int lane = threadIdx.x & 31;
  const int mofs = (lane >> 4) << 3;
#pragma unroll
  for (int tt = 0; tt < 4; ++tt) {
    const int nn = n0 + (tt << 4) + (lane & 15);
#pragma unroll
    for (int r = 0; r < 8; ++r) {
      const int mm = m0 + r + mofs;
      const __bf16 bv = (__bf16)((acc[tt][r] > 0.8f) ? 1.0f : 0.0f);
      mask[((size_t)(b * NPIX + mm) << 11) + nn] = bv;   // [B][1024][2048]
      maskT[((size_t)(b * NCTX + nn) << 10) + mm] = bv;  // [B][2048][1024]
    }
  }
}

// ---------------- Phase 2: degrees ----------------
__global__ void edge_deg(const __bf16* __restrict__ mask, float* __restrict__ dege_inv) {
  const int row = blockIdx.x * 4 + (threadIdx.x >> 5);  // 0..4095 (= b*1024+m)
  const int lane = threadIdx.x & 31;
  const __bf16* p = mask + ((size_t)row << 11);
  float s = 0.f;
  for (int i = lane; i < NCTX; i += 32) s += (float)p[i];
#pragma unroll
  for (int off = 16; off; off >>= 1) s += __shfl_down(s, off, 32);
  if (lane == 0) dege_inv[row] = 1.0f / (1.0f + s);  // +1 for the identity edge member
}

__global__ void v_deg(const __bf16* __restrict__ maskT, float* __restrict__ degv_inv) {
  const int row = blockIdx.x * 4 + (threadIdx.x >> 5);  // 0..8191 (= b*2048+m)
  const int lane = threadIdx.x & 31;
  const __bf16* p = maskT + ((size_t)row << 10);
  float s = 0.f;
  for (int i = lane; i < NPIX; i += 32) s += (float)p[i];
#pragma unroll
  for (int off = 16; off; off >>= 1) s += __shfl_down(s, off, 32);
  if (lane == 0) degv_inv[row] = 1.0f / fmaxf(s, 1.0f);
}

// ---------------- Generic linear: Out = In @ W^T + bias ----------------
__global__ void linear_wmma(const __bf16* __restrict__ In, const __bf16* __restrict__ Wb,
                            const float* __restrict__ bias, float* __restrict__ OutF,
                            __bf16* __restrict__ OutB, __bf16* __restrict__ OutT,
                            int rowsTot) {
  const int wave = blockIdx.x * (blockDim.x >> 5) + (threadIdx.x >> 5);
  const int m0 = (wave >> 4) << 4;
  const int n0 = (wave & 15) << 4;
  v8f acc = {};
  for (int k0 = 0; k0 < CDIM; k0 += 32) {
    v16bf a = frag_a(In, CDIM, m0, k0);
    v16bf bb = frag_bT(Wb, CDIM, n0, (size_t)k0);  // B[k][n] = W[n][k]
    acc = wmma_bf16(a, bb, acc);
  }
  const int lane = threadIdx.x & 31;
  const int nn = n0 + (lane & 15);
  const float bv = bias[nn];
  const int mofs = (lane >> 4) << 3;
#pragma unroll
  for (int r = 0; r < 8; ++r) {
    const int mm = m0 + r + mofs;
    const float v = acc[r] + bv;
    OutF[((size_t)mm << 8) + nn] = v;
    OutB[((size_t)mm << 8) + nn] = (__bf16)v;
    OutT[(size_t)nn * rowsTot + mm] = (__bf16)v;   // transposed copy for downstream B frags
  }
}

// ---------------- Phase 3: edge aggregation  Xe = (selfXn + mask@ctxXn) * dege_inv ----------------
// Block = 64x64 output; B panel (XnT ctx cols) staged in LDS. blocks/batch = 16*4 = 64.
__global__ void edge_agg(const __bf16* __restrict__ mask, const __bf16* __restrict__ XnT,
                         const float* __restrict__ Xn_f, const float* __restrict__ dege_inv,
                         __bf16* __restrict__ Xe_bf) {
  __shared__ __bf16 bpan[2][64 * 32];
  const int blk = blockIdx.x;
  const int b = blk >> 6;
  const int t = blk & 63;
  const int m0 = ((t >> 2) << 6) + ((threadIdx.x >> 5) << 4);
  const int n0 = (t & 3) << 6;
  const __bf16* A = mask + ((size_t)b << 21);          // [1024][2048]
  const size_t rowbase = (size_t)(b * VBLK + NPIX);    // ctx rows of Xn (X_all order)
  v8f acc[4] = {{}, {}, {}, {}};
  stage_b_panel(XnT, (size_t)NALL, n0, rowbase, bpan[0]);
#pragma unroll 1
  for (int i = 0; i < 64; ++i) {         // K = 2048
    const int k0 = i << 5;
    stage_fence();
    if (i + 1 < 64) stage_b_panel(XnT, (size_t)NALL, n0, rowbase + k0 + 32, bpan[(i + 1) & 1]);
    const v16bf a = frag_a(A, NCTX, m0, k0);
    const __bf16* P = bpan[i & 1];
#pragma unroll
    for (int tt = 0; tt < 4; ++tt) {
      const v16bf bb = frag_b_lds(P, tt << 4);
      acc[tt] = wmma_bf16(a, bb, acc[tt]);
    }
  }
  const int lane = threadIdx.x & 31;
  const int mofs = (lane >> 4) << 3;
#pragma unroll
  for (int tt = 0; tt < 4; ++tt) {
    const int nn = n0 + (tt << 4) + (lane & 15);
#pragma unroll
    for (int r = 0; r < 8; ++r) {
      const int mm = m0 + r + mofs;
      // faithful row mismatch: self node row = b*3072 + m in X_all ordering
      const float self = Xn_f[((size_t)(b * VBLK + mm) << 8) + nn];
      const float v = (acc[tt][r] + self) * dege_inv[b * NPIX + mm];
      Xe_bf[((size_t)(b * NPIX + mm) << 8) + nn] = (__bf16)v;
    }
  }
}

// ---------------- Phase 4: node aggregation  Xv_ctx = (maskT @ Xet) * degv_inv ----------------
// Block = 64x64 output; B panel (XetT cols) staged in LDS. blocks/batch = 32*4 = 128.
__global__ void v_agg(const __bf16* __restrict__ maskT, const __bf16* __restrict__ XetT,
                      const float* __restrict__ degv_inv, float* __restrict__ Xv) {
  __shared__ __bf16 bpan[2][64 * 32];
  const int blk = blockIdx.x;
  const int b = blk >> 7;
  const int t = blk & 127;
  const int m0 = ((t >> 2) << 6) + ((threadIdx.x >> 5) << 4);
  const int n0 = (t & 3) << 6;
  const __bf16* A = maskT + ((size_t)b << 21);   // [2048][1024]
  const size_t rowbase = (size_t)(b * NPIX);     // Xet rows of this batch
  v8f acc[4] = {{}, {}, {}, {}};
  stage_b_panel(XetT, (size_t)NTGT, n0, rowbase, bpan[0]);
#pragma unroll 1
  for (int i = 0; i < 32; ++i) {         // K = 1024
    const int k0 = i << 5;
    stage_fence();
    if (i + 1 < 32) stage_b_panel(XetT, (size_t)NTGT, n0, rowbase + k0 + 32, bpan[(i + 1) & 1]);
    const v16bf a = frag_a(A, NPIX, m0, k0);
    const __bf16* P = bpan[i & 1];
#pragma unroll
    for (int tt = 0; tt < 4; ++tt) {
      const v16bf bb = frag_b_lds(P, tt << 4);
      acc[tt] = wmma_bf16(a, bb, acc[tt]);
    }
  }
  const int lane = threadIdx.x & 31;
  const int mofs = (lane >> 4) << 3;
#pragma unroll
  for (int tt = 0; tt < 4; ++tt) {
    const int nn = n0 + (tt << 4) + (lane & 15);
#pragma unroll
    for (int r = 0; r < 8; ++r) {
      const int mm = m0 + r + mofs;
      Xv[((size_t)(b * VBLK + NPIX + mm) << 8) + nn] = acc[tt][r] * degv_inv[b * NCTX + mm];
    }
  }
}

// Identity-edge rows of Xv are plain copies of Xet.
__global__ void v_copy(const float* __restrict__ Xet_f, float* __restrict__ Xv) {
  const int idx = blockIdx.x * blockDim.x + threadIdx.x;  // 4096*256
  const int row = idx >> 8, cc = idx & 255;
  const int b = row >> 10, j = row & 1023;
  Xv[((size_t)(b * VBLK + j) << 8) + cc] = Xet_f[idx];
}

// ---------------- Phase 5: scatter Xv (H_big row order) to the 3 NCHW outputs ----------------
__global__ void scatter_out(const float* __restrict__ Xv, float* __restrict__ out) {
  const int idx = blockIdx.x * blockDim.x + threadIdx.x;  // 3 * 1048576
  const int which = idx >> 20;
  const int rem = idx & 1048575;
  const int b = rem >> 18;
  const int cch = (rem >> 10) & 255;
  const int n = rem & 1023;
  const int vrow = (which == 0) ? (b * NPIX + n)
                                : (NTGT + b * NCTX + (which - 1) * NPIX + n);
  out[idx] = Xv[((size_t)vrow << 8) + cch];
}

// ---------------- Host launch ----------------
extern "C" void kernel_launch(void* const* d_in, const int* in_sizes, int n_in,
                              void* d_out, int out_size, void* d_ws, size_t ws_size,
                              hipStream_t stream) {
  (void)in_sizes; (void)n_in; (void)out_size; (void)ws_size;
  const float* Xt  = (const float*)d_in[0];
  const float* Xc1 = (const float*)d_in[1];
  const float* Xc2 = (const float*)d_in[2];
  const float* Wn2e = (const float*)d_in[3];
  const float* bn2e = (const float*)d_in[4];
  const float* We2n = (const float*)d_in[5];
  const float* be2n = (const float*)d_in[6];
  float* out = (float*)d_out;

  char* ws = (char*)d_ws;
  size_t off = 0;
  auto take = [&](size_t bytes) { char* p = ws + off; off += (bytes + 255) & ~(size_t)255; return p; };
  __bf16* XA_bf   = (__bf16*)take((size_t)NALL * CDIM * 2);
  __bf16* NRM_bf  = (__bf16*)take((size_t)NALL * CDIM * 2);
  __bf16* mask    = (__bf16*)take((size_t)BNUM * NPIX * NCTX * 2);
  __bf16* maskT   = (__bf16*)take((size_t)BNUM * NCTX * NPIX * 2);
  __bf16* Wn2e_bf = (__bf16*)take((size_t)CDIM * CDIM * 2);
  __bf16* We2n_bf = (__bf16*)take((size_t)CDIM * CDIM * 2);
  float*  dege    = (float*)take((size_t)NTGT * 4);
  float*  degv    = (float*)take((size_t)NCTXALL * 4);
  float*  Xn_f    = (float*)take((size_t)NALL * CDIM * 4);
  __bf16* Xn_bf   = (__bf16*)take((size_t)NALL * CDIM * 2);
  __bf16* XnT_bf  = (__bf16*)take((size_t)NALL * CDIM * 2);   // [256][12288]
  __bf16* Xe_bf   = (__bf16*)take((size_t)NTGT * CDIM * 2);
  float*  Xet_f   = (float*)take((size_t)NTGT * CDIM * 4);
  __bf16* Xet_bf  = (__bf16*)take((size_t)NTGT * CDIM * 2);
  __bf16* XetT_bf = (__bf16*)take((size_t)NTGT * CDIM * 2);   // [256][4096]
  float*  Xv      = (float*)take((size_t)NALL * CDIM * 4);

  row_prep<<<NALL, CDIM, 0, stream>>>(Xt, Xc1, Xc2, XA_bf, NRM_bf);
  cvt_w<<<256, 256, 0, stream>>>(Wn2e, Wn2e_bf, CDIM * CDIM);
  cvt_w<<<256, 256, 0, stream>>>(We2n, We2n_bf, CDIM * CDIM);

  sim_mask<<<2048, 128, 0, stream>>>(NRM_bf, mask, maskT);
  edge_deg<<<1024, 128, 0, stream>>>(mask, dege);
  v_deg<<<2048, 128, 0, stream>>>(maskT, degv);

  linear_wmma<<<3072, 128, 0, stream>>>(XA_bf, Wn2e_bf, bn2e, Xn_f, Xn_bf, XnT_bf, NALL);
  edge_agg<<<256, 128, 0, stream>>>(mask, XnT_bf, Xn_f, dege, Xe_bf);
  linear_wmma<<<1024, 128, 0, stream>>>(Xe_bf, We2n_bf, be2n, Xet_f, Xet_bf, XetT_bf, NTGT);
  v_agg<<<512, 128, 0, stream>>>(maskT, XetT_bf, degv, Xv);
  v_copy<<<4096, 256, 0, stream>>>(Xet_f, Xv);

  scatter_out<<<12288, 256, 0, stream>>>(Xv, out);
}
